// QNetwork_6648609375076
// MI455X (gfx1250) — matrix operand order repaired and checked by reading, not compiled, not run on hardware
//
#include <hip/hip_runtime.h>
#include <hip/hip_bf16.h>
#include <math.h>

// Problem constants (match reference)
constexpr int B_ = 8, N_ = 5000, E_ = 80000;
constexpr int NODE_IN = 32, EDGE_IN = 16, HIDDEN = 128, EMBED = 64;
constexpr int BN = B_ * N_;     // 40000 nodes total
constexpr int BE = B_ * E_;     // 640000 edges total
constexpr int MLP_K = 4 * EMBED + EDGE_IN;   // 272
constexpr int MLP_KP = 288;                  // padded to 9 * 32

typedef __attribute__((ext_vector_type(16))) _Float16 v16h;
typedef __attribute__((ext_vector_type(8)))  _Float16 v8h;
typedef __attribute__((ext_vector_type(8)))  float    v8f;

// ---------------------------------------------------------------------------
// WMMA fragment helpers (CDNA5 ISA 7.12.2 layouts, wave32)
//
// A: 16x32 f16 from row-major. lanes 0-15 -> M=lane, lanes 16-31 -> M=lane-16.
// Per lane, halves a[0..7] = row[8*hi .. 8*hi+7], a[8..15] = row[16+8*hi ..]:
// two contiguous 16-byte chunks -> 2x b128 loads.
__device__ inline v16h load_A_frag(const _Float16* A, int ldA) {
  int lane = threadIdx.x & 31;
  int row  = lane & 15;
  int hi   = (lane >> 4) & 1;
  const _Float16* rp = A + (size_t)row * ldA + hi * 8;
  v8h lo = *(const v8h*)rp;
  v8h hh = *(const v8h*)(rp + 16);
  v16h a;
#pragma unroll
  for (int i = 0; i < 8; ++i) { a[i] = lo[i]; a[8 + i] = hh[i]; }
  return a;
}

// B fragments are pre-swizzled by k_prep into per-lane contiguous 32-byte
// records: layout [kt][nt][lane][16 halves]. One aligned v16h load per frag.
__device__ inline v16h load_B_packed(const _Float16* P, int kt, int nt, int NT) {
  int lane = threadIdx.x & 31;
  return *(const v16h*)(P + (((size_t)(kt * NT + nt) * 32 + lane) << 4));
}

// Pack one output half-element i of a B-fragment buffer from row-major W[K x ldN].
// i decomposes as [kt][nt][lane][h]; half h of lane = W[((h&7)+8*hiHalf+16*(h>>3))+32*kt][nt*16+lane%16]
__device__ inline void pack_B(const float* W, _Float16* out, int i,
                              int NT, int Kact, int ldN) {
  int h    = i & 15;
  int lane = (i >> 4) & 31;
  int nt   = (i >> 9) % NT;
  int kt   = (i >> 9) / NT;
  int kl   = (h & 7) + 16 * (h >> 3) + ((lane >= 16) ? 8 : 0);
  int k    = kt * 32 + kl;
  int n    = nt * 16 + (lane & 15);
  out[i] = (k < Kact) ? (_Float16)W[(size_t)k * ldN + n] : (_Float16)0.0f;
}

// C/D: 16x16 f32, VGPR r: lanes 0-15 -> (M=r, N=lane), lanes 16-31 -> (M=r+8, N=lane-16)
__device__ inline void store_D(float* D, int ldD, v8f c) {
  int lane = threadIdx.x & 31;
  int col  = lane & 15;
  int rb   = (lane >> 4) ? 8 : 0;
#pragma unroll
  for (int r = 0; r < 8; ++r) D[(size_t)(r + rb) * ldD + col] = c[r];
}

__device__ inline void atomicMaxFloat(float* addr, float val) {
  if (val >= 0.0f) atomicMax((int*)addr, __float_as_int(val));
  else             atomicMin((unsigned int*)addr, __float_as_uint(val));
}

// ---------------------------------------------------------------------------
// Init: zero/neg-inf all accumulators, q = b2
__global__ void k_init(float* agg0, float* agg1, float* m0, float* m1,
                       float* den0, float* den1, float* meanp, float* maxp,
                       float* q, const float* b2) {
  int i = blockIdx.x * blockDim.x + threadIdx.x;
  if (i < BN * HIDDEN) agg0[i] = 0.0f;
  if (i < BN * EMBED)  agg1[i] = 0.0f;
  if (i < BN) { m0[i] = -INFINITY; m1[i] = -INFINITY; den0[i] = 0.0f; den1[i] = 0.0f; }
  if (i < B_ * EMBED) { meanp[i] = 0.0f; maxp[i] = -INFINITY; }
  if (i < BE) q[i] = b2[0];
}

// Weight prep: pre-swizzled f16 B-fragment packs + folded edge vectors we = We @ ae
__global__ void k_prep(const float* W0, const float* W1g, const float* Wm,
                       const float* We0, const float* ae0,
                       const float* We1, const float* ae1,
                       _Float16* pg0, _Float16* pg1, _Float16* pmlp,
                       float* we0, float* we1) {
  int i = blockIdx.x * blockDim.x + threadIdx.x;
  if (i < NODE_IN * HIDDEN) pack_B(W0,  pg0,  i, 8, NODE_IN, HIDDEN);  // 1 kt x 8 nt
  if (i < HIDDEN * EMBED)   pack_B(W1g, pg1,  i, 4, HIDDEN,  EMBED);   // 4 kt x 4 nt
  if (i < MLP_KP * HIDDEN)  pack_B(Wm,  pmlp, i, 8, MLP_K,   HIDDEN);  // 9 kt x 8 nt
  if (i < EDGE_IN) {
    float acc = 0.0f;
    for (int j = 0; j < HIDDEN; ++j) acc += We0[i * HIDDEN + j] * ae0[j];
    we0[i] = acc;
  } else if (i < 2 * EDGE_IN) {
    int r = i - EDGE_IN; float acc = 0.0f;
    for (int j = 0; j < EMBED; ++j) acc += We1[r * EMBED + j] * ae1[j];
    we1[r] = acc;
  }
}

// LayerNorm nodes: one wave32 per 32-feature row
__global__ void k_ln_node(const float* x, const float* g, const float* b, _Float16* out) {
  int gt = blockIdx.x * blockDim.x + threadIdx.x;
  int row = gt >> 5, lane = gt & 31;
  float v = x[(size_t)row * NODE_IN + lane];
  float s = v;
  for (int m = 16; m >= 1; m >>= 1) s += __shfl_xor(s, m, 32);
  float mu = s * (1.0f / 32.0f);
  float d = v - mu;
  float s2 = d * d;
  for (int m = 16; m >= 1; m >>= 1) s2 += __shfl_xor(s2, m, 32);
  float rstd = rsqrtf(s2 * (1.0f / 32.0f) + 1e-5f);
  out[(size_t)row * NODE_IN + lane] = (_Float16)(d * rstd * g[lane] + b[lane]);
}

// LayerNorm edges (16 lanes/row) + fused per-edge projection scalars ef@(We·ae)
__global__ void k_ln_edge(const float* x, const float* g, const float* b,
                          const float* we0, const float* we1,
                          _Float16* out, float* ep0, float* ep1) {
  int gt = blockIdx.x * blockDim.x + threadIdx.x;
  int row = gt >> 4, lane = gt & 15;
  float v = x[(size_t)row * EDGE_IN + lane];
  float s = v;
  for (int m = 8; m >= 1; m >>= 1) s += __shfl_xor(s, m, 16);
  float mu = s * (1.0f / 16.0f);
  float d = v - mu;
  float s2 = d * d;
  for (int m = 8; m >= 1; m >>= 1) s2 += __shfl_xor(s2, m, 16);
  float rstd = rsqrtf(s2 * (1.0f / 16.0f) + 1e-5f);
  float nv = d * rstd * g[lane] + b[lane];
  out[(size_t)row * EDGE_IN + lane] = (_Float16)nv;
  float p0 = nv * we0[lane], p1 = nv * we1[lane];
  for (int m = 8; m >= 1; m >>= 1) { p0 += __shfl_xor(p0, m, 16); p1 += __shfl_xor(p1, m, 16); }
  if (lane == 0) { ep0[row] = p0; ep1[row] = p1; }
}

// h0 = nx[40000x32] @ W0[32x128]: one 16-row tile per block, 8 waves = 8 N-tiles, K=32 (1 WMMA)
__global__ void k_gemm_h0(const _Float16* A, const _Float16* Bp, float* H) {
  int mt = blockIdx.x;
  int w  = threadIdx.x >> 5;
  v16h a = load_A_frag(A + (size_t)mt * 16 * NODE_IN, NODE_IN);
  v16h b = load_B_packed(Bp, 0, w, 8);
  v8f c = {};
  c = __builtin_amdgcn_wmma_f32_16x16x32_f16(false, a, false, b, (short)0, c, false, false);
  store_D(H + (size_t)mt * 16 * HIDDEN + w * 16, HIDDEN, c);
}

// h1 = x1[40000x128] @ W1[128x64]: 4 waves = 4 N-tiles, K-loop of 4 WMMAs
__global__ void k_gemm_h1(const _Float16* A, const _Float16* Bp, float* H) {
  int mt = blockIdx.x;
  int w  = threadIdx.x >> 5;
  v8f c = {};
#pragma unroll
  for (int kt = 0; kt < 4; ++kt) {
    v16h a = load_A_frag(A + (size_t)mt * 16 * HIDDEN + kt * 32, HIDDEN);
    v16h b = load_B_packed(Bp, kt, w, 4);
    c = __builtin_amdgcn_wmma_f32_16x16x32_f16(false, a, false, b, (short)0, c, false, false);
  }
  store_D(H + (size_t)mt * 16 * EMBED + w * 16, EMBED, c);
}

// Per-node attention scalars s = h·asrc, d = h·adst (one wave per node)
template <int F>
__global__ void k_scalars(const float* h, const float* as, const float* ad,
                          float* s, float* d) {
  int gt = blockIdx.x * blockDim.x + threadIdx.x;
  int n = gt >> 5, lane = gt & 31;
  float ss = 0.0f, dd = 0.0f;
#pragma unroll
  for (int i = 0; i < F / 32; ++i) {
    int f = lane + i * 32;
    float hv = h[(size_t)n * F + f];
    ss += hv * as[f]; dd += hv * ad[f];
  }
  for (int m = 16; m >= 1; m >>= 1) { ss += __shfl_xor(ss, m, 32); dd += __shfl_xor(dd, m, 32); }
  if (lane == 0) { s[n] = ss; d[n] = dd; }
}

// logits = leaky_relu(s[src]+d[dst]+eproj, 0.2); segment-max into m[dst]
__global__ void k_logits(const int* ei, const float* s, const float* d,
                         const float* ep, float* logits, float* m) {
  int e = blockIdx.x * blockDim.x + threadIdx.x;
  int g = e / E_, le = e - g * E_;
  int src = ei[le] + g * N_;
  int dst = ei[E_ + le] + g * N_;
  float l = s[src] + d[dst] + ep[e];
  l = (l > 0.0f) ? l : 0.2f * l;
  logits[e] = l;
  atomicMaxFloat(&m[dst], l);
}

// ex = exp(l - m[dst]); segment-sum into denom[dst]  (in-place over logits)
__global__ void k_exp(const int* ei, float* logits, const float* m, float* den) {
  int e = blockIdx.x * blockDim.x + threadIdx.x;
  int g = e / E_, le = e - g * E_;
  int dst = ei[E_ + le] + g * N_;
  float ex = expf(logits[e] - m[dst]);
  logits[e] = ex;
  atomicAdd(&den[dst], ex);
}

// agg[dst] += alpha * h[src] (one wave per edge, F/32 features per lane)
template <int F>
__global__ void k_agg(const int* ei, const float* ex, const float* den,
                      const float* h, float* agg) {
  int gt = blockIdx.x * blockDim.x + threadIdx.x;
  int e = gt >> 5, lane = gt & 31;
  int g = e / E_, le = e - g * E_;
  int src = ei[le] + g * N_;
  int dst = ei[E_ + le] + g * N_;
  float alpha = ex[e] / (den[dst] + 1e-16f);
#pragma unroll
  for (int i = 0; i < F / 32; ++i) {
    int f = lane + i * 32;
    atomicAdd(&agg[(size_t)dst * F + f], alpha * h[(size_t)src * F + f]);
  }
}

// x1 = elu(agg0 + b0)  (fp32 + f16 copy for next GEMM)
__global__ void k_finish0(const float* agg, const float* b, float* x1, _Float16* x1h) {
  int i = blockIdx.x * blockDim.x + threadIdx.x;
  int f = i & (HIDDEN - 1);
  float v = agg[i] + b[f];
  v = (v > 0.0f) ? v : expm1f(v);
  x1[i] = v;
  x1h[i] = (_Float16)v;
}

// node_emb = agg1 + b1; fused mean/max pooling into per-graph context
__global__ void k_finish1(const float* agg, const float* b, float* emb, _Float16* embh,
                          float* meanp, float* maxp) {
  int i = blockIdx.x * blockDim.x + threadIdx.x;
  int n = i >> 6, f = i & (EMBED - 1);
  float v = agg[i] + b[f];
  emb[i] = v;
  embh[i] = (_Float16)v;
  int g = n / N_;
  atomicAdd(&meanp[g * EMBED + f], v * (1.0f / (float)N_));
  atomicMaxFloat(&maxp[g * EMBED + f], v);
}

// ctx[b] = concat(mean, max) -> f16
__global__ void k_ctx(const float* meanp, const float* maxp, _Float16* ctx) {
  int i = blockIdx.x * blockDim.x + threadIdx.x;
  int g = i >> 7, f = i & 127;
  float v = (f < EMBED) ? meanp[g * EMBED + f] : maxp[g * EMBED + f - EMBED];
  ctx[i] = (_Float16)v;
}

// Per-edge MLP: gather-concat 16x288 A-tile into LDS in 16-byte chunks
// (all concat segments are multiples of 8 halves), 8 waves x 9 WMMAs,
// fused bias+ReLU+(·W2) epilogue with half-wave shfl reduction.
__global__ void k_mlp(const int* ei, const _Float16* emb, const _Float16* ef,
                      const _Float16* ctx, const _Float16* Bp, const float* b1,
                      const float* W2, float* q) {
  __shared__ _Float16 tA[16 * MLP_KP];
  __shared__ int ssrc[16], sdst[16], sg[16];
  int tid = threadIdx.x;
  int e0 = blockIdx.x * 16;
  if (tid < 16) {
    int e = e0 + tid;
    int g = e / E_, le = e - g * E_;
    ssrc[tid] = ei[le] + g * N_;
    sdst[tid] = ei[E_ + le] + g * N_;
    sg[tid] = g;
  }
  __syncthreads();
  // 16 rows x 36 chunks of 8 halves (16B each)
  for (int idx = tid; idx < 16 * (MLP_KP / 8); idx += 256) {
    int row = idx / (MLP_KP / 8);
    int col = (idx - row * (MLP_KP / 8)) * 8;
    v8h v = {};
    if (col < EMBED)                     v = *(const v8h*)(emb + (size_t)ssrc[row] * EMBED + col);
    else if (col < 2 * EMBED)            v = *(const v8h*)(emb + (size_t)sdst[row] * EMBED + (col - EMBED));
    else if (col < 2 * EMBED + EDGE_IN)  v = *(const v8h*)(ef + (size_t)(e0 + row) * EDGE_IN + (col - 2 * EMBED));
    else if (col < MLP_K)                v = *(const v8h*)(ctx + (size_t)sg[row] * (2 * EMBED) + (col - 2 * EMBED - EDGE_IN));
    *(v8h*)(tA + row * MLP_KP + col) = v;
  }
  __syncthreads();
  int w = tid >> 5, lane = tid & 31;
  v8f c = {};
#pragma unroll
  for (int kt = 0; kt < MLP_KP / 32; ++kt) {
    v16h a = load_A_frag(tA + kt * 32, MLP_KP);
    v16h b = load_B_packed(Bp, kt, w, 8);
    c = __builtin_amdgcn_wmma_f32_16x16x32_f16(false, a, false, b, (short)0, c, false, false);
  }
  int col = w * 16 + (lane & 15);
  float bb = b1[col], ww = W2[col];
  int rb = (lane >= 16) ? 8 : 0;
#pragma unroll
  for (int r = 0; r < 8; ++r) {
    float v = c[r] + bb;
    v = (v > 0.0f) ? v : 0.0f;
    v *= ww;
    for (int m = 8; m >= 1; m >>= 1) v += __shfl_xor(v, m, 16);
    if ((lane & 15) == 0) atomicAdd(&q[e0 + rb + r], v);
  }
}

// ---------------------------------------------------------------------------
extern "C" void kernel_launch(void* const* d_in, const int* in_sizes, int n_in,
                              void* d_out, int out_size, void* d_ws, size_t ws_size,
                              hipStream_t stream) {
  const float* node_x    = (const float*)d_in[0];
  const float* edge_attr = (const float*)d_in[1];
  const int*   ei        = (const int*)  d_in[2];
  const float* ln_ng = (const float*)d_in[3];
  const float* ln_nb = (const float*)d_in[4];
  const float* ln_eg = (const float*)d_in[5];
  const float* ln_eb = (const float*)d_in[6];
  const float* g0W  = (const float*)d_in[7];
  const float* g0We = (const float*)d_in[8];
  const float* g0as = (const float*)d_in[9];
  const float* g0ad = (const float*)d_in[10];
  const float* g0ae = (const float*)d_in[11];
  const float* g0b  = (const float*)d_in[12];
  const float* g1W  = (const float*)d_in[13];
  const float* g1We = (const float*)d_in[14];
  const float* g1as = (const float*)d_in[15];
  const float* g1ad = (const float*)d_in[16];
  const float* g1ae = (const float*)d_in[17];
  const float* g1b  = (const float*)d_in[18];
  const float* mW1  = (const float*)d_in[19];
  const float* mb1  = (const float*)d_in[20];
  const float* mW2  = (const float*)d_in[21];
  const float* mb2  = (const float*)d_in[22];
  float* q = (float*)d_out;

  char* ws = (char*)d_ws;
  size_t off = 0;
  auto alloc = [&](size_t bytes) -> void* {
    void* p = ws + off;
    off += (bytes + 255) & ~(size_t)255;
    return p;
  };

  _Float16* nxh   = (_Float16*)alloc((size_t)BN * NODE_IN * 2);
  _Float16* efh   = (_Float16*)alloc((size_t)BE * EDGE_IN * 2);
  float*    ep0   = (float*)   alloc((size_t)BE * 4);
  float*    ep1   = (float*)   alloc((size_t)BE * 4);
  _Float16* pg0   = (_Float16*)alloc((size_t)NODE_IN * HIDDEN * 2);
  _Float16* pg1   = (_Float16*)alloc((size_t)HIDDEN * EMBED * 2);
  _Float16* pmlp  = (_Float16*)alloc((size_t)MLP_KP * HIDDEN * 2);
  float*    we0   = (float*)   alloc(EDGE_IN * 4);
  float*    we1   = (float*)   alloc(EDGE_IN * 4);
  float*    h0    = (float*)   alloc((size_t)BN * HIDDEN * 4);
  float*    s0    = (float*)   alloc((size_t)BN * 4);
  float*    d0    = (float*)   alloc((size_t)BN * 4);
  float*    m0    = (float*)   alloc((size_t)BN * 4);
  float*    den0  = (float*)   alloc((size_t)BN * 4);
  float*    logits= (float*)   alloc((size_t)BE * 4);
  float*    agg0  = (float*)   alloc((size_t)BN * HIDDEN * 4);
  float*    x1    = (float*)   alloc((size_t)BN * HIDDEN * 4);
  _Float16* x1h   = (_Float16*)alloc((size_t)BN * HIDDEN * 2);
  float*    h1    = (float*)   alloc((size_t)BN * EMBED * 4);
  float*    s1    = (float*)   alloc((size_t)BN * 4);
  float*    d1    = (float*)   alloc((size_t)BN * 4);
  float*    m1    = (float*)   alloc((size_t)BN * 4);
  float*    den1  = (float*)   alloc((size_t)BN * 4);
  float*    agg1  = (float*)   alloc((size_t)BN * EMBED * 4);
  float*    emb   = (float*)   alloc((size_t)BN * EMBED * 4);
  _Float16* embh  = (_Float16*)alloc((size_t)BN * EMBED * 2);
  float*    meanp = (float*)   alloc((size_t)B_ * EMBED * 4);
  float*    maxp  = (float*)   alloc((size_t)B_ * EMBED * 4);
  _Float16* ctx   = (_Float16*)alloc((size_t)B_ * 2 * EMBED * 2);

  (void)in_sizes; (void)n_in; (void)out_size; (void)ws_size;

  // 0. init accumulators + q = b2
  k_init<<<(BN * HIDDEN + 255) / 256, 256, 0, stream>>>(
      agg0, agg1, m0, m1, den0, den1, meanp, maxp, q, mb2);
  // 1. weight prep (B-fragment pre-swizzle)
  k_prep<<<(MLP_KP * HIDDEN + 255) / 256, 256, 0, stream>>>(
      g0W, g1W, mW1, g0We, g0ae, g1We, g1ae, pg0, pg1, pmlp, we0, we1);
  // 2. layernorms (+ fused edge projections)
  k_ln_node<<<BN * 32 / 256, 256, 0, stream>>>(node_x, ln_ng, ln_nb, nxh);
  k_ln_edge<<<BE * 16 / 256, 256, 0, stream>>>(edge_attr, ln_eg, ln_eb, we0, we1,
                                               efh, ep0, ep1);
  // 3. GAT layer 0
  k_gemm_h0<<<BN / 16, 256, 0, stream>>>(nxh, pg0, h0);
  k_scalars<HIDDEN><<<BN * 32 / 256, 256, 0, stream>>>(h0, g0as, g0ad, s0, d0);
  k_logits<<<BE / 256, 256, 0, stream>>>(ei, s0, d0, ep0, logits, m0);
  k_exp<<<BE / 256, 256, 0, stream>>>(ei, logits, m0, den0);
  k_agg<HIDDEN><<<BE * 32 / 256, 256, 0, stream>>>(ei, logits, den0, h0, agg0);
  k_finish0<<<BN * HIDDEN / 256, 256, 0, stream>>>(agg0, g0b, x1, x1h);
  // 4. GAT layer 1
  k_gemm_h1<<<BN / 16, 128, 0, stream>>>(x1h, pg1, h1);
  k_scalars<EMBED><<<BN * 32 / 256, 256, 0, stream>>>(h1, g1as, g1ad, s1, d1);
  k_logits<<<BE / 256, 256, 0, stream>>>(ei, s1, d1, ep1, logits, m1);
  k_exp<<<BE / 256, 256, 0, stream>>>(ei, logits, m1, den1);
  k_agg<EMBED><<<BE * 32 / 256, 256, 0, stream>>>(ei, logits, den1, h1, agg1);
  k_finish1<<<BN * EMBED / 256, 256, 0, stream>>>(agg1, g1b, emb, embh, meanp, maxp);
  // 5. context + per-edge MLP (WMMA, dominant 45 GFLOP)
  k_ctx<<<B_ * 128 / 256, 256, 0, stream>>>(meanp, maxp, ctx);
  k_mlp<<<BE / 16, 256, 0, stream>>>(ei, embh, efh, ctx, pmlp, mb1, mW2, q);
}
